// LiquidOperator_48387101556761
// MI455X (gfx1250) — matrix operand order repaired and checked by reading, not compiled, not run on hardware
//
#include <hip/hip_runtime.h>

// ---------------- problem constants ----------------
#define LEVELS   17
#define UNITS    51
#define VARS     4
#define BLEN     32768
#define PREDN    4
#define TLEN     (BLEN + PREDN)   // 32772
#define DFEAT    68
#define UNFOLDS  6
#define EPSV     1e-8f
#define M1DIM    42               // (17+68)/2
#define GDIM     272              // 4*68
#define G2DIM    170              // (272+68)/2

typedef _Float16 v16h __attribute__((ext_vector_type(16)));
typedef float    v8f  __attribute__((ext_vector_type(8)));

// ---- pre-swizzled weight-fragment region (one fragment = 512 halves = 1KB)
// fragment index within a matrix = kc * NT + nt ; each fragment is the exact
// per-lane WMMA B-operand register image: lane L holds col n = nt*16 + (L&15),
// K = kc*32 + (L>>4)*8 + (e&7) + (e>>3)*16  (e = half index 0..15). K/N pads are ZERO.
#define FRAG_HALVES 512
#define FR_PW1   0            // pre_w1: 68x68  -> 3 kc x 5 nt = 15
#define FR_PW2   15           // pre_w2: 68x68  -> 15
#define FR_TW1   30           // tW1: 17x17     -> 1 x 2 = 2
#define FR_TW2   32           // tW2: 17x42     -> 1 x 3 = 3
#define FR_TW3   35           // tW3: 42x68     -> 2 x 5 = 10
#define FR_GW1   45           // gW1: 272x272   -> 9 x 17 = 153
#define FR_GW2   198          // gW2: 272x170   -> 9 x 11 = 99
#define FR_GW3   297          // gW3: 170x68    -> 6 x 5 = 30
#define FR_TOTAL 327
#define FRAG_BYTES ((size_t)FR_TOTAL * FRAG_HALVES * sizeof(_Float16))  // 334848

// ---- zero-padded bias region (floats, offsets within bias pad area)
#define BP_B1   0      // 80
#define BP_TB1  80     // 32
#define BP_TB2  112    // 48
#define BP_TB3  160    // 80
#define BP_GB1  240    // 272
#define BP_GB2  512    // 176
#define BP_TOTAL 688
#define BP_BYTES ((size_t)1024 * sizeof(float))

// fast sigmoid: v_exp_f32 + v_rcp_f32 (no IEEE div fixup chain)
__device__ __forceinline__ float sigf(float x) {
    return __builtin_amdgcn_rcpf(1.0f + __expf(-x));
}
__device__ __forceinline__ float softplusf(float x) {
    return (x > 20.0f) ? x : log1pf(__expf(x));
}

// ---------------- WMMA fragment helpers (wave32, 16x16x32 f16 -> f32) -------
// A fragment: 16x32 f16 tile from LDS (row-major, leading dim ld).
// ISA 7.12.2: lane L holds row M=L&15; lane group g=L>>4 supplies K offset g*8;
// VGPR v holds K = (v>>2)*16 + g*8 + (v&3)*2 (+0/+1).
__device__ __forceinline__ v16h load_a(const _Float16* __restrict__ src,
                                       int ld, int kbase, int lane) {
    v16h a;
    int m  = lane & 15;
    int g8 = (lane >> 4) * 8;
    #pragma unroll
    for (int v = 0; v < 8; ++v) {
        int k = ((v >> 2) * 16) + g8 + ((v & 3) * 2);
        a[2 * v]     = src[m * ld + kbase + k];
        a[2 * v + 1] = src[m * ld + kbase + k + 1];
    }
    return a;
}

// One 16x16 output tile of act(16xK) @ W, W pre-swizzled into fragments.
__device__ __forceinline__ v8f tile_mm(const _Float16* __restrict__ A, int ald,
                                       const _Float16* __restrict__ fragbase,
                                       int NT, int nt, int nkc, int lane) {
    v8f c = {};
    for (int kc = 0; kc < nkc; ++kc) {
        v16h a = load_a(A, ald, kc * 32, lane);
        v16h b = *(const v16h*)(fragbase +
                  ((size_t)(kc * NT + nt) * 32 + lane) * 16);
        c = __builtin_amdgcn_wmma_f32_16x16x32_f16(
                false, a, false, b, (short)0, c, false, false);
    }
    return c;
}

// Unconditional hidden-layer store: bias is zero-padded, pad columns receive
// finite sigmoid values which the next layer's zero K-pad rows annihilate.
// C/D layout: VGPR r, lane L -> row = (L>>4)*8 + r, col = L&15.
__device__ __forceinline__ void store_tile(v8f c, _Float16* __restrict__ D, int dld,
                                           int noff, int nbase,
                                           const float* __restrict__ biaspad,
                                           int lane) {
    int col   = lane & 15;
    int rbase = (lane >> 4) * 8;
    int n     = nbase + col;
    float bv  = biaspad[n];
    #pragma unroll
    for (int r = 0; r < 8; ++r) {
        float x = sigf(c[r] + bv);
        D[(rbase + r) * dld + noff + n] = (_Float16)x;
    }
}

__device__ __forceinline__ void zero_h(_Float16* __restrict__ p, int n, int lane) {
    for (int i = lane; i < n; i += 32) p[i] = (_Float16)0.0f;
}

// ---------------- kernel 0a: weight prep (one wave per fragment) ------------
__global__ __launch_bounds__(32) void prep_kernel(const float* __restrict__ W,
                                                  int K, int N, int NT,
                                                  _Float16* __restrict__ out) {
    int lane = threadIdx.x & 31;
    int f  = blockIdx.x;
    int kc = f / NT, nt = f % NT;
    int n  = nt * 16 + (lane & 15);
    int k0 = kc * 32 + (lane >> 4) * 8;
    v16h b;
    #pragma unroll
    for (int e = 0; e < 16; ++e) {
        int k = k0 + (e & 7) + (e >> 3) * 16;
        b[e] = (k < K && n < N) ? (_Float16)W[(size_t)k * N + n] : (_Float16)0.0f;
    }
    *(v16h*)(out + ((size_t)f * 32 + lane) * 16) = b;
}

// ---------------- kernel 0b: bias prep (zero-pad) ---------------------------
__global__ void bias_prep_kernel(const float* __restrict__ b, int N,
                                 float* __restrict__ out) {
    int i = threadIdx.x;
    out[i] = (i < N) ? b[i] : 0.0f;
}

// ---------------- kernel 1: VarEncoder -> seq[v][t][17] ---------------------
__global__ __launch_bounds__(32) void encoder_kernel(
    const float* __restrict__ x,
    const float* __restrict__ bpad,   // padded b1 at BP_B1
    const float* __restrict__ b2,
    const _Float16* __restrict__ frags,
    float* __restrict__ seq)
{
    __shared__ _Float16 bufA[16 * 96];
    __shared__ _Float16 bufB[16 * 96];
    int lane = threadIdx.x & 31;
    int row0 = blockIdx.x * 16;

    if (row0 + 16 < BLEN)
        __builtin_prefetch(&x[(size_t)(row0 + 16) * DFEAT + lane * 34], 0, 1);

    zero_h(bufA, 16 * 96, lane);
    zero_h(bufB, 16 * 96, lane);
    __syncthreads();
    for (int i = lane; i < 16 * DFEAT; i += 32) {
        int r = i / DFEAT, cidx = i % DFEAT;
        bufA[r * 96 + cidx] = (_Float16)x[(size_t)(row0 + r) * DFEAT + cidx];
    }
    __syncthreads();

    // layer 1 (sigmoid)
    const _Float16* fW1 = frags + (size_t)FR_PW1 * FRAG_HALVES;
    for (int nt = 0; nt < 5; ++nt) {
        v8f c = tile_mm(bufA, 96, fW1, 5, nt, 3, lane);
        store_tile(c, bufB, 96, 0, nt * 16, bpad + BP_B1, lane);
    }
    __syncthreads();

    // layer 2 (no sigmoid) -> scatter into per-variable sequences
    const _Float16* fW2 = frags + (size_t)FR_PW2 * FRAG_HALVES;
    for (int nt = 0; nt < 5; ++nt) {
        v8f c = tile_mm(bufB, 96, fW2, 5, nt, 3, lane);
        int col = lane & 15, rb = (lane >> 4) * 8;
        int n = nt * 16 + col;
        if (n < DFEAT) {
            float bv = b2[n];
            int v = n / LEVELS, s = n % LEVELS;
            #pragma unroll
            for (int r = 0; r < 8; ++r) {
                int t = row0 + rb + r;
                seq[((size_t)v * BLEN + t) * LEVELS + s] = c[r] + bv;
            }
        }
    }
}

// ---------------- kernel 2: LTC scan (serial in t, 4 blocks) ----------------
__global__ __launch_bounds__(64) void scan_kernel(
    const float* __restrict__ seq,
    const float* __restrict__ gleak, const float* __restrict__ vleak,
    const float* __restrict__ cm,
    const float* __restrict__ w, const float* __restrict__ sigma,
    const float* __restrict__ mu, const float* __restrict__ erev,
    const float* __restrict__ sens_w, const float* __restrict__ sens_sigma,
    const float* __restrict__ sens_mu, const float* __restrict__ sens_erev,
    const float* __restrict__ in_w, const float* __restrict__ in_b,
    const float* __restrict__ out_w, const float* __restrict__ out_b,
    const float* __restrict__ mask, const float* __restrict__ smask,
    float* __restrict__ rbuf)
{
    __shared__ float l_wp[UNITS * UNITS], l_mu[UNITS * UNITS];
    __shared__ float l_sg[UNITS * UNITS], l_er[UNITS * UNITS];
    __shared__ float l_swp[LEVELS * UNITS], l_smu[LEVELS * UNITS];
    __shared__ float l_ssg[LEVELS * UNITS], l_ser[LEVELS * UNITS];
    __shared__ float vst[UNITS];
    __shared__ float ubuf[LEVELS];

    int tid = threadIdx.x;
    int vb  = blockIdx.x;

    const float* wv  = w     + (size_t)vb * UNITS * UNITS;
    const float* sgv = sigma + (size_t)vb * UNITS * UNITS;
    const float* muv = mu    + (size_t)vb * UNITS * UNITS;
    const float* erv = erev  + (size_t)vb * UNITS * UNITS;
    const float* swv = sens_w     + (size_t)vb * LEVELS * UNITS;
    const float* ssv = sens_sigma + (size_t)vb * LEVELS * UNITS;
    const float* smv = sens_mu    + (size_t)vb * LEVELS * UNITS;
    const float* sev = sens_erev  + (size_t)vb * LEVELS * UNITS;

    for (int i = tid; i < UNITS * UNITS; i += 64) {
        l_wp[i] = softplusf(wv[i]) * mask[i];
        l_mu[i] = muv[i];
        l_sg[i] = sgv[i];
        l_er[i] = erv[i];
    }
    for (int i = tid; i < LEVELS * UNITS; i += 64) {
        l_swp[i] = softplusf(swv[i]) * smask[i];
        l_smu[i] = smv[i];
        l_ssg[i] = ssv[i];
        l_ser[i] = sev[i];
    }
    if (tid < UNITS) vst[tid] = 0.0f;

    float glj = 0.f, glvlj = 0.f, cmtj = 0.f;
    if (tid < UNITS) {
        glj   = softplusf(gleak[vb * UNITS + tid]);
        glvlj = glj * vleak[vb * UNITS + tid];
        cmtj  = softplusf(cm[vb * UNITS + tid]) * (float)UNFOLDS;
    }
    float iw = 0.f, ib = 0.f, ow = 0.f, ob = 0.f;
    if (tid < LEVELS) {
        iw = in_w[vb * LEVELS + tid];
        ib = in_b[vb * LEVELS + tid];
        ow = out_w[vb * LEVELS + tid];
        ob = out_b[vb * LEVELS + tid];
    }
    __syncthreads();

    for (int t = 0; t < TLEN; ++t) {
        int ts = (t < BLEN) ? t : (BLEN - 1);
        if (tid < LEVELS)
            ubuf[tid] = seq[((size_t)vb * BLEN + ts) * LEVELS + tid] * iw + ib;
        __syncthreads();

        float nsum = 0.f, dsum = 0.f;
        if (tid < UNITS) {
            for (int s = 0; s < LEVELS; ++s) {
                int idx = s * UNITS + tid;
                float sa = l_swp[idx] * sigf((ubuf[s] - l_smu[idx]) * l_ssg[idx]);
                nsum += sa * l_ser[idx];
                dsum += sa;
            }
        }

        for (int u = 0; u < UNFOLDS; ++u) {
            float vv = 0.f;
            if (tid < UNITS) {
                float num = 0.f, den = 0.f;
                float vj = vst[tid];
                for (int i = 0; i < UNITS; ++i) {
                    int idx = i * UNITS + tid;
                    float wa = l_wp[idx] * sigf((vst[i] - l_mu[idx]) * l_sg[idx]);
                    num += wa * l_er[idx];
                    den += wa;
                }
                vv = (cmtj * vj + glvlj + num + nsum) *
                     __builtin_amdgcn_rcpf(cmtj + glj + den + dsum + EPSV);
            }
            __syncthreads();
            if (tid < UNITS) vst[tid] = vv;
            __syncthreads();
        }

        if (tid < LEVELS)
            rbuf[(size_t)t * DFEAT + vb * LEVELS + tid] = vst[tid] * ow + ob;
        __syncthreads();
    }
}

// ---------------- kernel 3: CrossConnector ----------------------------------
__global__ __launch_bounds__(32) void cross_kernel(
    const float* __restrict__ rbuf,
    const float* __restrict__ bpad,    // padded tb1/tb2/tb3/gb1/gb2
    const float* __restrict__ gb3,
    const _Float16* __restrict__ frags,
    float* __restrict__ out)
{
    __shared__ _Float16 sA[16 * 64];
    __shared__ _Float16 sB[16 * 64];
    __shared__ _Float16 tbuf[16 * 288];   // (T,272) tile, K-padded to 288
    __shared__ _Float16 g1b[16 * 288];
    __shared__ _Float16 g2b[16 * 192];    // 170 padded to 192

    int lane = threadIdx.x & 31;
    int row0 = blockIdx.x * 16;

    if (row0 + 16 < TLEN)
        __builtin_prefetch(&rbuf[(size_t)(row0 + 16) * DFEAT + lane * 34], 0, 1);

    const _Float16* fT1 = frags + (size_t)FR_TW1 * FRAG_HALVES;
    const _Float16* fT2 = frags + (size_t)FR_TW2 * FRAG_HALVES;
    const _Float16* fT3 = frags + (size_t)FR_TW3 * FRAG_HALVES;
    const _Float16* fG1 = frags + (size_t)FR_GW1 * FRAG_HALVES;
    const _Float16* fG2 = frags + (size_t)FR_GW2 * FRAG_HALVES;
    const _Float16* fG3 = frags + (size_t)FR_GW3 * FRAG_HALVES;

    // one-time zeroing: every later pad write is a finite sigmoid value,
    // and all weight fragments have zero K-pad rows.
    zero_h(sA, 16 * 64, lane);
    zero_h(sB, 16 * 64, lane);
    zero_h(tbuf, 16 * 288, lane);
    zero_h(g1b, 16 * 288, lane);
    zero_h(g2b, 16 * 192, lane);
    __syncthreads();

    // per-variable tile MLP: 17 -> 17 -> 42 -> 68, all sigmoid
    for (int v = 0; v < VARS; ++v) {
        for (int i = lane; i < 16 * LEVELS; i += 32) {
            int r = i / LEVELS, cidx = i % LEVELS;
            int t = row0 + r;
            sA[r * 64 + cidx] =
                (t < TLEN) ? (_Float16)rbuf[(size_t)t * DFEAT + v * LEVELS + cidx]
                           : (_Float16)0.0f;
        }
        __syncthreads();

        for (int nt = 0; nt < 2; ++nt) {
            v8f c = tile_mm(sA, 64, fT1, 2, nt, 1, lane);
            store_tile(c, sB, 64, 0, nt * 16, bpad + BP_TB1, lane);
        }
        __syncthreads();

        for (int nt = 0; nt < 3; ++nt) {
            v8f c = tile_mm(sB, 64, fT2, 3, nt, 1, lane);
            store_tile(c, sA, 64, 0, nt * 16, bpad + BP_TB2, lane);
        }
        __syncthreads();

        for (int nt = 0; nt < 5; ++nt) {
            v8f c = tile_mm(sA, 64, fT3, 5, nt, 2, lane);
            store_tile(c, tbuf, 288, v * DFEAT, nt * 16, bpad + BP_TB3, lane);
        }
        __syncthreads();
    }

    // g MLP: 272 -> 272 (sig) -> 170 (sig) -> 68
    for (int nt = 0; nt < 17; ++nt) {
        v8f c = tile_mm(tbuf, 288, fG1, 17, nt, 9, lane);
        store_tile(c, g1b, 288, 0, nt * 16, bpad + BP_GB1, lane);
    }
    __syncthreads();

    for (int nt = 0; nt < 11; ++nt) {
        v8f c = tile_mm(g1b, 288, fG2, 11, nt, 9, lane);
        store_tile(c, g2b, 192, 0, nt * 16, bpad + BP_GB2, lane);
    }
    __syncthreads();

    for (int nt = 0; nt < 5; ++nt) {
        v8f c = tile_mm(g2b, 192, fG3, 5, nt, 6, lane);
        int col = lane & 15, rb = (lane >> 4) * 8;
        int n = nt * 16 + col;
        if (n < DFEAT) {
            float bv = gb3[n];
            #pragma unroll
            for (int r = 0; r < 8; ++r) {
                int t = row0 + rb + r;
                if (t < TLEN) out[(size_t)t * DFEAT + n] = c[r] + bv;
            }
        }
    }
}

// ---------------- launcher ---------------------------------------------------
extern "C" void kernel_launch(void* const* d_in, const int* in_sizes, int n_in,
                              void* d_out, int out_size, void* d_ws, size_t ws_size,
                              hipStream_t stream) {
    (void)in_sizes; (void)n_in; (void)out_size; (void)ws_size;

    const float* x      = (const float*)d_in[0];
    const float* pre_w1 = (const float*)d_in[1];
    const float* pre_b1 = (const float*)d_in[2];
    const float* pre_w2 = (const float*)d_in[3];
    const float* pre_b2 = (const float*)d_in[4];
    const float* gleak  = (const float*)d_in[5];
    const float* vleak  = (const float*)d_in[6];
    const float* cm     = (const float*)d_in[7];
    const float* w      = (const float*)d_in[8];
    const float* sigma  = (const float*)d_in[9];
    const float* mu     = (const float*)d_in[10];
    const float* erev   = (const float*)d_in[11];
    const float* sens_w = (const float*)d_in[12];
    const float* sens_sigma = (const float*)d_in[13];
    const float* sens_mu    = (const float*)d_in[14];
    const float* sens_erev  = (const float*)d_in[15];
    const float* in_w  = (const float*)d_in[16];
    const float* in_b  = (const float*)d_in[17];
    const float* out_w = (const float*)d_in[18];
    const float* out_b = (const float*)d_in[19];
    const float* tW1 = (const float*)d_in[20];
    const float* tb1 = (const float*)d_in[21];
    const float* tW2 = (const float*)d_in[22];
    const float* tb2 = (const float*)d_in[23];
    const float* tW3 = (const float*)d_in[24];
    const float* tb3 = (const float*)d_in[25];
    const float* gW1 = (const float*)d_in[26];
    const float* gb1 = (const float*)d_in[27];
    const float* gW2 = (const float*)d_in[28];
    const float* gb2 = (const float*)d_in[29];
    const float* gW3 = (const float*)d_in[30];
    const float* gb3 = (const float*)d_in[31];
    const float* mask  = (const float*)d_in[32];
    const float* smask = (const float*)d_in[33];

    // ws layout: [weight fragments f16][padded biases f32][seq f32][rbuf f32]
    _Float16* frags = (_Float16*)d_ws;
    char* p = (char*)d_ws + ((FRAG_BYTES + 1023) & ~(size_t)1023);
    float* bpad = (float*)p;
    p += BP_BYTES;
    float* seq  = (float*)p;                                 // (4, 32768, 17)
    float* rbuf = seq + (size_t)VARS * BLEN * LEVELS;        // (32772, 68)
    float* outp = (float*)d_out;                             // (32772, 68)

    // one-time weight swizzle into WMMA B-fragment order
    prep_kernel<<<15, 32, 0, stream>>>(pre_w1, DFEAT, DFEAT, 5,
                                       frags + (size_t)FR_PW1 * FRAG_HALVES);
    prep_kernel<<<15, 32, 0, stream>>>(pre_w2, DFEAT, DFEAT, 5,
                                       frags + (size_t)FR_PW2 * FRAG_HALVES);
    prep_kernel<<<2, 32, 0, stream>>>(tW1, LEVELS, LEVELS, 2,
                                      frags + (size_t)FR_TW1 * FRAG_HALVES);
    prep_kernel<<<3, 32, 0, stream>>>(tW2, LEVELS, M1DIM, 3,
                                      frags + (size_t)FR_TW2 * FRAG_HALVES);
    prep_kernel<<<10, 32, 0, stream>>>(tW3, M1DIM, DFEAT, 5,
                                       frags + (size_t)FR_TW3 * FRAG_HALVES);
    prep_kernel<<<153, 32, 0, stream>>>(gW1, GDIM, GDIM, 17,
                                        frags + (size_t)FR_GW1 * FRAG_HALVES);
    prep_kernel<<<99, 32, 0, stream>>>(gW2, GDIM, G2DIM, 11,
                                       frags + (size_t)FR_GW2 * FRAG_HALVES);
    prep_kernel<<<30, 32, 0, stream>>>(gW3, G2DIM, DFEAT, 5,
                                       frags + (size_t)FR_GW3 * FRAG_HALVES);

    // zero-padded biases
    bias_prep_kernel<<<1, 80,  0, stream>>>(pre_b1, DFEAT,  bpad + BP_B1);
    bias_prep_kernel<<<1, 32,  0, stream>>>(tb1,    LEVELS, bpad + BP_TB1);
    bias_prep_kernel<<<1, 48,  0, stream>>>(tb2,    M1DIM,  bpad + BP_TB2);
    bias_prep_kernel<<<1, 80,  0, stream>>>(tb3,    DFEAT,  bpad + BP_TB3);
    bias_prep_kernel<<<1, 272, 0, stream>>>(gb1,    GDIM,   bpad + BP_GB1);
    bias_prep_kernel<<<1, 176, 0, stream>>>(gb2,    G2DIM,  bpad + BP_GB2);

    encoder_kernel<<<BLEN / 16, 32, 0, stream>>>(x, bpad, pre_b2, frags, seq);

    scan_kernel<<<VARS, 64, 0, stream>>>(seq, gleak, vleak, cm, w, sigma, mu, erev,
                                         sens_w, sens_sigma, sens_mu, sens_erev,
                                         in_w, in_b, out_w, out_b, mask, smask, rbuf);

    cross_kernel<<<(TLEN + 15) / 16, 32, 0, stream>>>(rbuf, bpad, gb3, frags, outp);
}
